// MHSA_16080357556766
// MI455X (gfx1250) — matrix operand order repaired and checked by reading, not compiled, not run on hardware
//
#include <hip/hip_runtime.h>
#include <hip/hip_bf16.h>

typedef __attribute__((ext_vector_type(16))) __bf16 v16bf;
typedef __attribute__((ext_vector_type(8)))  __bf16 v8bf;
typedef __attribute__((ext_vector_type(8)))  float  v8f;

#define HEADS 8
#define DHEAD 128
#define LSEQ  1024
#define SCALEQ 0.08838834764831845f   // 128^-0.5

// ---------------- workspace layout (bytes) ----------------
constexpr size_t OFF_WBF = 0;                         // 3072x512 bf16
constexpr size_t SZ_WBF  = 3072ull * 512 * 2;
constexpr size_t OFF_XT  = OFF_WBF + SZ_WBF;          // (b,pix,c) bf16
constexpr size_t SZ_XT   = 8ull * 1024 * 512 * 2;
constexpr size_t OFF_Q   = OFF_XT + SZ_XT;            // (b,h,L,d) bf16, pre-scaled
constexpr size_t SZ_QKV  = 8ull * HEADS * LSEQ * DHEAD * 2;
constexpr size_t OFF_K   = OFF_Q + SZ_QKV;            // (b,h,L,d) bf16
constexpr size_t OFF_VT  = OFF_K + SZ_QKV;            // (b,h,d,L) bf16
constexpr size_t OFF_RW  = OFF_VT + SZ_QKV;           // 64x128 bf16 (row 63 = 0)
constexpr size_t OFF_RH  = OFF_RW + 64 * 128 * 2;     // 64x128 bf16

// ---------------- helpers ----------------
__device__ inline v8f wmma_bf16(v16bf a, v16bf b, v8f c) {
  return __builtin_amdgcn_wmma_f32_16x16x32_bf16(false, a, false, b, (short)0, c,
                                                 false, false);
}

__device__ inline v16bf ld16(const __bf16* p0, const __bf16* p1) {
  v8bf a = *(const v8bf*)p0;
  v8bf b = *(const v8bf*)p1;
  v16bf r;
#pragma unroll
  for (int i = 0; i < 8; ++i) { r[i] = a[i]; r[i + 8] = b[i]; }
  return r;
}

__device__ inline float hmax16(float v) {
#pragma unroll
  for (int m = 8; m >= 1; m >>= 1) v = fmaxf(v, __shfl_xor(v, m, 32));
  return v;  // reduction stays inside the 16-lane half that owns a row
}
__device__ inline float hsum16(float v) {
#pragma unroll
  for (int m = 8; m >= 1; m >>= 1) v += __shfl_xor(v, m, 32);
  return v;
}
__device__ inline v8f zero8() {
  v8f z = {0.f, 0.f, 0.f, 0.f, 0.f, 0.f, 0.f, 0.f};
  return z;
}

// ---------------- prep kernels ----------------
__global__ void cvt_w_kernel(const float* __restrict__ w, __bf16* __restrict__ wbf,
                             int n) {
  int i = blockIdx.x * blockDim.x + threadIdx.x;
  if (i < n) wbf[i] = (__bf16)w[i];
}

// featuremap (b, c=512, p=1024) -> xt (b, p, c) bf16
__global__ void transpose_x_kernel(const float* __restrict__ x,
                                   __bf16* __restrict__ xt) {
  int i = blockIdx.x * blockDim.x + threadIdx.x;  // over 8*512*1024
  int b = i >> 19;
  int r = i & ((1 << 19) - 1);
  int c = r >> 10;
  int p = r & 1023;
  xt[((size_t)b * 1024 + p) * 512 + c] = (__bf16)x[i];
}

__global__ void prep_rel_kernel(const float* __restrict__ rh,
                                const float* __restrict__ rw,
                                __bf16* __restrict__ rhp,
                                __bf16* __restrict__ rwp) {
  int i = blockIdx.x * blockDim.x + threadIdx.x;  // 64*128
  int r = i >> 7;
  int c = i & 127;
  rhp[i] = (r < 63) ? (__bf16)rh[r * 128 + c] : (__bf16)0.f;
  rwp[i] = (r < 63) ? (__bf16)rw[r * 128 + c] : (__bf16)0.f;
}

// ---------------- QKV projection: qkv[o,p] = sum_c W[o,c]*X[c,p] ----------------
// grid (16 Ntiles, 24 Mtiles, 8 batch), block 256 (8 waves)
__global__ __launch_bounds__(256) void qkv_proj_kernel(
    const __bf16* __restrict__ wbf, const __bf16* __restrict__ xt,
    __bf16* __restrict__ qws, __bf16* __restrict__ kws, __bf16* __restrict__ vtws) {
  int nb = blockIdx.x, mb = blockIdx.y, b = blockIdx.z;
  int w = threadIdx.x >> 5, l = threadIdx.x & 31, l16 = l & 15, hi = l >> 4;

  int orow = mb * 128 + w * 16 + l16;  // A-fragment row for this lane
  const __bf16* arow = wbf + (size_t)orow * 512;
  int pixc = nb * 64;

  v8f acc[4];
#pragma unroll
  for (int ct = 0; ct < 4; ++ct) acc[ct] = zero8();

  for (int ks = 0; ks < 16; ++ks) {
    int kb = ks * 32;
    v16bf a = ld16(arow + kb + hi * 16, arow + kb + hi * 16 + 8);
#pragma unroll
    for (int ct = 0; ct < 4; ++ct) {
      const __bf16* bp =
          xt + ((size_t)b * 1024 + pixc + ct * 16 + l16) * 512 + kb + hi * 16;
      acc[ct] = wmma_bf16(a, *(const v16bf*)bp, acc[ct]);
    }
  }

  int obase = mb * 128 + w * 16 + 8 * hi;  // o for fragment element jj=0
  int sec = obase >> 10;                   // 0=Q 1=K 2=V
  int head = (obase >> 7) & 7;
  int f0 = obase & 127;

#pragma unroll
  for (int ct = 0; ct < 4; ++ct) {
    int pix = pixc + ct * 16 + l16;
    if (sec == 2) {
#pragma unroll
      for (int jj = 0; jj < 8; ++jj)
        vtws[(((size_t)b * HEADS + head) * DHEAD + f0 + jj) * LSEQ + pix] =
            (__bf16)acc[ct][jj];
    } else {
      float sc = (sec == 0) ? SCALEQ : 1.0f;
      v8bf pk;
#pragma unroll
      for (int jj = 0; jj < 8; ++jj) pk[jj] = (__bf16)(acc[ct][jj] * sc);
      __bf16* dst = (sec == 0 ? qws : kws) +
                    (((size_t)b * HEADS + head) * LSEQ + pix) * DHEAD + f0;
      *(v8bf*)dst = pk;
    }
  }
}

// ---------------- fused flash attention with relative logits ----------------
// grid (8 qblocks, 8 heads, 8 batch), block 256: wave w owns rows qblk*128+w*16..+15
__global__ __launch_bounds__(256) void attn_kernel(
    const __bf16* __restrict__ qws, const __bf16* __restrict__ kws,
    const __bf16* __restrict__ vtws, const __bf16* __restrict__ rwp,
    const __bf16* __restrict__ rhp, float* __restrict__ out) {
  __shared__ __attribute__((aligned(32))) __bf16 pwlds[8][16][64];
  __shared__ __attribute__((aligned(32))) __bf16 phlds[8][16][64];
  __shared__ __attribute__((aligned(32))) __bf16 pstage[8][16][64];

  int qblk = blockIdx.x, head = blockIdx.y, b = blockIdx.z;
  int w = threadIdx.x >> 5, l = threadIdx.x & 31, l16 = l & 15, hi = l >> 4;
  size_t bh = (size_t)b * HEADS + head;
  const __bf16* Q = qws + bh * LSEQ * DHEAD;
  const __bf16* K = kws + bh * LSEQ * DHEAD;
  const __bf16* V = vtws + bh * DHEAD * LSEQ;
  int qrow0 = qblk * 128 + w * 16;

  // Persistent Q A-fragments for this wave's 16 rows (4 k-steps of 32)
  const __bf16* qrow = Q + (size_t)(qrow0 + l16) * DHEAD;
  v16bf qa[4];
#pragma unroll
  for (int ks = 0; ks < 4; ++ks)
    qa[ks] = ld16(qrow + ks * 32 + hi * 16, qrow + ks * 32 + hi * 16 + 8);

  // Rel-pos tables: Pw[x,r]=q_x.relW[r], Ph[x,r]=q_x.relH[r]  (r padded to 64)
#pragma unroll
  for (int ct = 0; ct < 4; ++ct) {
    v8f pw = zero8(), ph = zero8();
    const __bf16* rwrow = rwp + (ct * 16 + l16) * 128;
    const __bf16* rhrow = rhp + (ct * 16 + l16) * 128;
#pragma unroll
    for (int ks = 0; ks < 4; ++ks) {
      pw = wmma_bf16(qa[ks], *(const v16bf*)(rwrow + ks * 32 + hi * 16), pw);
      ph = wmma_bf16(qa[ks], *(const v16bf*)(rhrow + ks * 32 + hi * 16), ph);
    }
#pragma unroll
    for (int jj = 0; jj < 8; ++jj) {
      pwlds[w][jj + 8 * hi][ct * 16 + l16] = (__bf16)pw[jj];
      phlds[w][jj + 8 * hi][ct * 16 + l16] = (__bf16)ph[jj];
    }
  }
  __syncthreads();

  float row_m[8], row_l[8];
  v8f acc[8];
#pragma unroll
  for (int jj = 0; jj < 8; ++jj) { row_m[jj] = -3.0e38f; row_l[jj] = 0.f; }
#pragma unroll
  for (int o = 0; o < 8; ++o) acc[o] = zero8();

  for (int j = 0; j < 16; ++j) {  // key tiles of 64
    int y0 = j * 64;

    // S = Q * K^T  (4 col-tiles x 4 k-steps)
    v8f s[4];
#pragma unroll
    for (int ct = 0; ct < 4; ++ct) {
      v8f sv = zero8();
      const __bf16* krow = K + (size_t)(y0 + ct * 16 + l16) * DHEAD;
#pragma unroll
      for (int ks = 0; ks < 4; ++ks)
        sv = wmma_bf16(qa[ks], *(const v16bf*)(krow + ks * 32 + hi * 16), sv);
      s[ct] = sv;
    }

    // + relative logits (shifted gather from the per-wave P tables)
#pragma unroll
    for (int ct = 0; ct < 4; ++ct) {
      int y = y0 + ct * 16 + l16;
      int wy = y & 31, hy = (y >> 5) & 31;
#pragma unroll
      for (int jj = 0; jj < 8; ++jj) {
        int xr = jj + 8 * hi;
        int qp = qrow0 + xr;
        int wx = qp & 31, hx = (qp >> 5) & 31;
        s[ct][jj] += (float)pwlds[w][xr][wy - wx + 31] +
                     (float)phlds[w][xr][hy - hx + 31];
      }
    }

    // Online softmax (row stats are wave-local: each lane-half owns 8 rows)
#pragma unroll
    for (int jj = 0; jj < 8; ++jj) {
      float tm = fmaxf(fmaxf(s[0][jj], s[1][jj]), fmaxf(s[2][jj], s[3][jj]));
      tm = hmax16(tm);
      float nm = fmaxf(row_m[jj], tm);
      float alpha = __expf(row_m[jj] - nm);
      row_m[jj] = nm;
      float rs = 0.f;
#pragma unroll
      for (int ct = 0; ct < 4; ++ct) {
        float p = __expf(s[ct][jj] - nm);
        s[ct][jj] = p;
        rs += p;
      }
      rs = hsum16(rs);
      row_l[jj] = row_l[jj] * alpha + rs;
#pragma unroll
      for (int o = 0; o < 8; ++o) acc[o][jj] *= alpha;
    }

    // Stage P (D-layout -> A-layout via LDS, wave-private region)
#pragma unroll
    for (int ct = 0; ct < 4; ++ct)
#pragma unroll
      for (int jj = 0; jj < 8; ++jj)
        pstage[w][jj + 8 * hi][ct * 16 + l16] = (__bf16)s[ct][jj];
    __syncthreads();

    const __bf16* pr = &pstage[w][l16][0];
    v16bf pa0 = ld16(pr + hi * 16, pr + hi * 16 + 8);
    v16bf pa1 = ld16(pr + 32 + hi * 16, pr + 40 + hi * 16);

    // O += P * V   (V stored transposed (d,L): B-fragments are contiguous)
#pragma unroll
    for (int o = 0; o < 8; ++o) {
      const __bf16* vrow = V + (size_t)(o * 16 + l16) * LSEQ + y0;
      acc[o] = wmma_bf16(pa0, *(const v16bf*)(vrow + hi * 16), acc[o]);
      acc[o] = wmma_bf16(pa1, *(const v16bf*)(vrow + 32 + hi * 16), acc[o]);
    }
    __syncthreads();
  }

  // Epilogue: divide by row sums, write (b, n*d, h, w) layout
  float* ob = out + bh * DHEAD * LSEQ;
#pragma unroll
  for (int jj = 0; jj < 8; ++jj) {
    float inv = 1.0f / row_l[jj];
    int qp = qrow0 + jj + 8 * hi;
#pragma unroll
    for (int o = 0; o < 8; ++o)
      ob[(size_t)(o * 16 + l16) * LSEQ + qp] = acc[o][jj] * inv;
  }
}

// ---------------- launch ----------------
extern "C" void kernel_launch(void* const* d_in, const int* in_sizes, int n_in,
                              void* d_out, int out_size, void* d_ws, size_t ws_size,
                              hipStream_t stream) {
  const float* feat = (const float*)d_in[0];   // (8,512,32,32)
  const float* wqkv = (const float*)d_in[1];   // (3072,512)
  const float* relh = (const float*)d_in[2];   // (63,128)
  const float* relw = (const float*)d_in[3];   // (63,128)
  float* out = (float*)d_out;                  // (8,1024,32,32) f32

  char* ws = (char*)d_ws;
  __bf16* wbf = (__bf16*)(ws + OFF_WBF);
  __bf16* xt  = (__bf16*)(ws + OFF_XT);
  __bf16* qws = (__bf16*)(ws + OFF_Q);
  __bf16* kws = (__bf16*)(ws + OFF_K);
  __bf16* vtw = (__bf16*)(ws + OFF_VT);
  __bf16* rwp = (__bf16*)(ws + OFF_RW);
  __bf16* rhp = (__bf16*)(ws + OFF_RH);

  cvt_w_kernel<<<(3072 * 512) / 256, 256, 0, stream>>>(wqkv, wbf, 3072 * 512);
  transpose_x_kernel<<<(8 * 512 * 1024) / 256, 256, 0, stream>>>(feat, xt);
  prep_rel_kernel<<<(64 * 128) / 256, 256, 0, stream>>>(relh, relw, rhp, rwp);

  qkv_proj_kernel<<<dim3(16, 24, 8), 256, 0, stream>>>(wbf, xt, qws, kws, vtw);
  attn_kernel<<<dim3(8, 8, 8), 256, 0, stream>>>(qws, kws, vtw, rwp, rhp, out);
}
